// SvegaSelfAttentionHead_5248450036559
// MI455X (gfx1250) — matrix operand order repaired and checked by reading, not compiled, not run on hardware
//
#include <hip/hip_runtime.h>
#include <math.h>

// ---------------------------------------------------------------------------
// Types for CDNA5 WMMA (wave32, 16x16x32 bf16 -> f32) and TDM descriptors
// ---------------------------------------------------------------------------
typedef __bf16 bf16_t;
typedef __attribute__((ext_vector_type(16))) __bf16        v16bf;
typedef __attribute__((ext_vector_type(8)))  __bf16        v8bf;
typedef __attribute__((ext_vector_type(8)))  float         v8f;
typedef __attribute__((ext_vector_type(4)))  unsigned int  v4u;
typedef __attribute__((ext_vector_type(8)))  int           v8i;
typedef __attribute__((ext_vector_type(4)))  int           v4i;

#define BATCH 8
#define SEQ   2048
#define EMB   1024
#define HD    64
#define SCALE 0.125f   // 1/sqrt(64)

static __device__ inline v8f vzero8() {
    v8f z;
#pragma unroll
    for (int i = 0; i < 8; ++i) z[i] = 0.0f;
    return z;
}

static __device__ inline v16bf cat16(v8bf a, v8bf b) {
    return __builtin_shufflevector(a, b, 0,1,2,3,4,5,6,7,8,9,10,11,12,13,14,15);
}

static __device__ inline v8f wmma_bf16(v16bf a, v16bf b, v8f c) {
    // D = A(16x32 bf16) * B(32x16 bf16) + C(16x16 f32)
    return __builtin_amdgcn_wmma_f32_16x16x32_bf16(
        /*neg_a=*/false, a, /*neg_b=*/false, b,
        /*c_mod=*/(short)0, c, /*reuse_a=*/false, /*reuse_b=*/false);
}

static __device__ inline float rmax16(float v) {
#pragma unroll
    for (int m = 1; m < 16; m <<= 1) v = fmaxf(v, __shfl_xor(v, m, 32));
    return v;
}
static __device__ inline float rsum16(float v) {
#pragma unroll
    for (int m = 1; m < 16; m <<= 1) v += __shfl_xor(v, m, 32);
    return v;
}

// Tensor Data Mover builtin arity differs across toolchains (bridge doc):
// ROCm 7.2 / clang-22: 5 args; amdgpu-toolchain / clang-23: 6 args.
#if defined(__clang_major__) && (__clang_major__ >= 23)
#define TDM_LOAD_TO_LDS(g0, g1, g2, g3, cpol)                                  \
    __builtin_amdgcn_tensor_load_to_lds((g0), (g1), (g2), (g3), (v8i)(0), (cpol))
#else
#define TDM_LOAD_TO_LDS(g0, g1, g2, g3, cpol)                                  \
    __builtin_amdgcn_tensor_load_to_lds((g0), (g1), (g2), (g3), (cpol))
#endif

// ---------------------------------------------------------------------------
// Kernel 0: one-time weight transpose+downconvert.
// W[e][d] fp32 (1024x64) -> Wt[d][e] bf16 (64x1024), so kernel 1's WMMA
// B-fragments become two contiguous 16B loads (no gather, no cvt in the loop).
// ---------------------------------------------------------------------------
__global__ __launch_bounds__(256)
void wt_transpose_kernel(const float* __restrict__ Wq,
                         const float* __restrict__ Wk,
                         const float* __restrict__ Wv,
                         bf16_t* __restrict__ Wt)   // [3][64][1024]
{
    const int idx = blockIdx.x * 256 + threadIdx.x;   // 0 .. 64*1024-1
    const int mat = blockIdx.y;                       // 0=Q 1=K 2=V
    const int d   = idx >> 10;                        // 0..63
    const int e   = idx & (EMB - 1);                  // 0..1023
    const float* W = (mat == 0) ? Wq : (mat == 1) ? Wk : Wv;
    Wt[((size_t)mat * HD + d) * EMB + e] = (__bf16)W[(size_t)e * HD + d];
}

// ---------------------------------------------------------------------------
// Kernel 1: QKV projection.  One block = 3 waves (q,k,v), one 16-row M tile.
// The 16x1024 fp32 X tile (64KB) is staged into LDS once per block by the
// Tensor Data Mover (TENSOR_LOAD_TO_LDS, waited with s_wait_tensorcnt), then
// all three waves build bf16 A-fragments from LDS and run 4 WMMAs per K-step.
// Q,K stored row-major bf16 [B*S][64]; V stored transposed bf16 [B][64][S].
// ---------------------------------------------------------------------------
__global__ __launch_bounds__(96)
void qkv_project_kernel(const float* __restrict__ X,
                        const bf16_t* __restrict__ Wt,   // [3][64][1024] bf16
                        bf16_t* __restrict__ Qb,
                        bf16_t* __restrict__ Kb,
                        bf16_t* __restrict__ Vt)
{
    __shared__ __align__(16) float lds_x[16 * EMB];   // 64 KB X tile

    const int lane  = threadIdx.x & 31;
    const int wave  = threadIdx.x >> 5;      // 0=Q, 1=K, 2=V
    const int mtile = blockIdx.x;            // 0 .. B*S/16-1
    const int col   = lane & 15;
    const int hi    = lane >> 4;
    const int offA  = hi ? 16 : 0;           // A K-split across lane halves
    const int offB  = hi ? 16 : 0;           // B K-split across lane halves

    // ---- TDM: DMA the 16x1024 fp32 tile of X into LDS (wave 0 only) ----
    if (wave == 0) {
        const unsigned           lds_base = (unsigned)(size_t)(void*)lds_x;
        const unsigned long long ga =
            (unsigned long long)(size_t)(X + (size_t)mtile * 16 * EMB);

        v4u g0;                               // D# group 0 (128b)
        g0[0] = 1u;                           // count=1 (valid user descriptor)
        g0[1] = lds_base;                     // lds_addr (bytes)
        g0[2] = (unsigned)(ga & 0xFFFFFFFFu);            // global_addr[31:0]
        g0[3] = (unsigned)((ga >> 32) & 0x01FFFFFFu)     // global_addr[56:32]
              | (2u << 30);                              // type = 2 ("image")

        v8i g1;                               // D# group 1 (256b)
        g1[0] = (int)(2u << 16);              // data_size = 2 -> 4-byte elems
        g1[1] = (int)((unsigned)(EMB & 0xFFFF) << 16);   // tensor_dim0[15:0]
        g1[2] = (int)(((unsigned)EMB >> 16) | (16u << 16)); // dim0 hi | tensor_dim1=16
        g1[3] = (int)((unsigned)(EMB & 0xFFFF) << 16);   // tile_dim0 = 1024
        g1[4] = 16;                           // tile_dim1 = 16 rows
        g1[5] = EMB;                          // tensor_dim0_stride[31:0] = 1024
        g1[6] = 0;                            // stride hi | tensor_dim1_stride lo
        g1[7] = 0;

        v4i z4;                               // groups 2/3 unused (2D tensor)
        z4[0] = z4[1] = z4[2] = z4[3] = 0;

        TDM_LOAD_TO_LDS(g0, g1, z4, z4, 0);
        __builtin_amdgcn_s_wait_tensorcnt(0);
    }
    __syncthreads();                          // publish LDS tile to all waves

    const float*  xr = lds_x + (size_t)col * EMB;      // this lane's A row
    const bf16_t* wb = Wt + (size_t)wave * HD * EMB;   // this wave's Wt

    v8f acc[4];
#pragma unroll
    for (int n = 0; n < 4; ++n) acc[n] = vzero8();

    for (int k0 = 0; k0 < EMB; k0 += 32) {
        // A fragment: 16x32 bf16 tile of X, built from the LDS stage
        v16bf a;
#pragma unroll
        for (int i = 0; i < 8; ++i) {
            a[i]     = (__bf16)xr[k0 + offA + i];
            a[8 + i] = (__bf16)xr[k0 + offA + 8 + i];
        }
#pragma unroll
        for (int n = 0; n < 4; ++n) {
            // B fragment: contiguous bf16 from the pre-transposed weights
            const bf16_t* wrow = wb + (size_t)(n * 16 + col) * EMB + k0 + offB;
            v16bf b = cat16(*(const v8bf*)wrow, *(const v8bf*)(wrow + 8));
            acc[n] = wmma_bf16(a, b, acc[n]);
        }
    }

    // Scatter accumulators per the 16x16 f32 C/D layout: VGPR j -> M = j + 8*hi.
    // Branch once per wave (wave-uniform) so the store loops stay exec-clean.
    if (wave < 2) {
        bf16_t* dst = (wave == 0) ? Qb : Kb;
#pragma unroll
        for (int n = 0; n < 4; ++n) {
#pragma unroll
            for (int j = 0; j < 8; ++j) {
                const int row = mtile * 16 + j + 8 * hi;
                dst[(size_t)row * HD + n * 16 + col] = (__bf16)acc[n][j];
            }
        }
    } else {
#pragma unroll
        for (int n = 0; n < 4; ++n) {
#pragma unroll
            for (int j = 0; j < 8; ++j) {
                const int row = mtile * 16 + j + 8 * hi;
                const int b   = row >> 11;        // row / SEQ
                const int s   = row & (SEQ - 1);
                Vt[((size_t)b * HD + n * 16 + col) * SEQ + s] = (__bf16)acc[n][j];
            }
        }
    }
}

// ---------------------------------------------------------------------------
// Kernel 2: flash attention, one wave per block, one 16-query tile per wave.
// Streams 32-key chunks: 4 WMMAs for scores, online softmax (16-lane shuffles),
// LDS bounce to transpose P from C-layout to A-layout, 4 WMMAs for P*V.
// ---------------------------------------------------------------------------
__global__ __launch_bounds__(32)
void flash_attn_kernel(const bf16_t* __restrict__ Qb,
                       const bf16_t* __restrict__ Kb,
                       const bf16_t* __restrict__ Vt,
                       float* __restrict__ out)
{
    __shared__ __align__(16) bf16_t pbuf[16 * 32];   // P tile bounce buffer

    const int lane  = threadIdx.x & 31;
    const int col   = lane & 15;
    const int hi    = lane >> 4;
    const int g     = blockIdx.x;            // q-tile id, 0..1023
    const int row0  = g * 16;                // global row base
    const int batch = row0 >> 11;
    const int q0    = row0 & (SEQ - 1);      // local query base within batch
    const int offA  = hi ? 16 : 0;
    const int offB  = hi ? 16 : 0;

    // Preload Q A-fragments for the two K=32 steps covering D=64.
    const bf16_t* qrow = Qb + (size_t)(row0 + col) * HD;
    v16bf qa[2];
#pragma unroll
    for (int t = 0; t < 2; ++t) {
        v8bf lo  = *(const v8bf*)(qrow + t * 32 + offA);
        v8bf hi8 = *(const v8bf*)(qrow + t * 32 + offA + 8);
        qa[t] = cat16(lo, hi8);
    }

    v8f acc[4];
#pragma unroll
    for (int n = 0; n < 4; ++n) acc[n] = vzero8();
    float m_s[8], l_s[8], alpha[8];
#pragma unroll
    for (int j = 0; j < 8; ++j) { m_s[j] = -3.0e38f; l_s[j] = 0.0f; }

    const int kmax = q0 + 15;                // causal: max key any row needs

    for (int k0 = 0; k0 <= kmax; k0 += 32) {
        // Prefetch next chunk's K rows into cache (speculative OK).
        __builtin_prefetch(Kb + (size_t)(batch * SEQ + k0 + 32 + col) * HD, 0, 1);

        // ---- scores: two 16-key tiles, each K=64 via two WMMAs ----
        v8f s0 = vzero8(), s1 = vzero8();
#pragma unroll
        for (int t = 0; t < 2; ++t) {
            const bf16_t* krow =
                Kb + (size_t)(batch * SEQ + k0 + t * 16 + col) * HD;
#pragma unroll
            for (int d = 0; d < 2; ++d) {
                v8bf b0 = *(const v8bf*)(krow + d * 32 + offB);
                v8bf b1 = *(const v8bf*)(krow + d * 32 + offB + 8);
                v16bf kb = cat16(b0, b1);
                if (t == 0) s0 = wmma_bf16(qa[d], kb, s0);
                else        s1 = wmma_bf16(qa[d], kb, s1);
            }
        }

        // ---- online softmax over this 32-key chunk ----
        const bool needMask = (k0 + 31 > q0);
#pragma unroll
        for (int j = 0; j < 8; ++j) {
            const int qrow_l = q0 + j + 8 * hi;
            float x0 = s0[j] * SCALE;
            float x1 = s1[j] * SCALE;
            if (needMask) {
                if (k0 + col      > qrow_l) x0 = -INFINITY;
                if (k0 + 16 + col > qrow_l) x1 = -INFINITY;
            }
            float cm   = rmax16(fmaxf(x0, x1));
            float mnew = fmaxf(m_s[j], cm);
            float al   = __expf(m_s[j] - mnew);
            m_s[j]     = mnew;
            float p0   = __expf(x0 - mnew);
            float p1   = __expf(x1 - mnew);
            l_s[j]     = l_s[j] * al + rsum16(p0 + p1);
            alpha[j]   = al;
            const int r = j + 8 * hi;
            pbuf[r * 32 + col]      = (__bf16)p0;   // C-layout -> row-major LDS
            pbuf[r * 32 + 16 + col] = (__bf16)p1;
        }

        // Wave-local LDS RAW fence before the transposed reads.
        asm volatile("s_wait_dscnt 0" ::: "memory");

        // Reload P as an A-fragment (16x32 bf16, row-major in LDS).
        v8bf p0v = *(const v8bf*)(pbuf + col * 32 + offA);
        v8bf p1v = *(const v8bf*)(pbuf + col * 32 + offA + 8);
        v16bf pa = cat16(p0v, p1v);

        // Rescale running output by alpha, then accumulate P*V.
#pragma unroll
        for (int n = 0; n < 4; ++n) {
#pragma unroll
            for (int j = 0; j < 8; ++j) acc[n][j] *= alpha[j];
        }
#pragma unroll
        for (int n = 0; n < 4; ++n) {
            const bf16_t* vp = Vt + ((size_t)batch * HD + n * 16 + col) * SEQ
                                  + k0 + offB;
            v16bf vb = cat16(*(const v8bf*)vp, *(const v8bf*)(vp + 8));
            acc[n] = wmma_bf16(pa, vb, acc[n]);
        }
    }

    // ---- epilogue: normalize and store fp32 output ----
#pragma unroll
    for (int n = 0; n < 4; ++n) {
#pragma unroll
        for (int j = 0; j < 8; ++j) {
            const int row = row0 + j + 8 * hi;
            out[(size_t)row * HD + n * 16 + col] = acc[n][j] / l_s[j];
        }
    }
}

// ---------------------------------------------------------------------------
// Launch.  ws layout (bf16): Q [B*S][64], K [B*S][64], Vt [B][64][S],
// Wt [3][64][1024]  -> ~6.4 MB total.
// ---------------------------------------------------------------------------
extern "C" void kernel_launch(void* const* d_in, const int* in_sizes, int n_in,
                              void* d_out, int out_size, void* d_ws, size_t ws_size,
                              hipStream_t stream) {
    const float* X  = (const float*)d_in[0];
    const float* Wq = (const float*)d_in[1];
    const float* Wk = (const float*)d_in[2];
    const float* Wv = (const float*)d_in[3];
    float* out = (float*)d_out;

    const size_t BSD = (size_t)BATCH * SEQ * HD;
    bf16_t* Qb = (bf16_t*)d_ws;
    bf16_t* Kb = Qb + BSD;
    bf16_t* Vt = Kb + BSD;
    bf16_t* Wt = Vt + BSD;                       // [3][64][1024]

    // One-time weight transpose/downconvert (tiny; W lives in L2 afterwards).
    wt_transpose_kernel<<<dim3((HD * EMB) / 256, 3), 256, 0, stream>>>(
        Wq, Wk, Wv, Wt);

    // 1024 M-tiles of 16 rows; 3 waves/block; TDM stages X tile into LDS.
    qkv_project_kernel<<<(BATCH * SEQ) / 16, 96, 0, stream>>>(
        X, Wt, Qb, Kb, Vt);

    // 1024 q-tiles, one wave each (small blocks spread the causal-length
    // imbalance across many independently-schedulable workgroups).
    flash_attn_kernel<<<(BATCH * SEQ) / 16, 32, 0, stream>>>(Qb, Kb, Vt, out);
}
